// MultiscaleDeformablePixelDecoder_84430467105178
// MI455X (gfx1250) — matrix operand order repaired and checked by reading, not compiled, not run on hardware
//
#include <hip/hip_runtime.h>

typedef __bf16 v16bf __attribute__((ext_vector_type(16)));
typedef __bf16 v8bf  __attribute__((ext_vector_type(8)));
typedef float  v8f   __attribute__((ext_vector_type(8)));

#define NQ      12096
#define NBATCH  2
#define ROWS    (NBATCH * NQ)   // 24192
#define DMODEL  256
#define MHEADS  8
#define DHEAD   32
#define NLVL    3
#define NPTS    4

// ---------------------------------------------------------------------------
// h = hidden + pos  ->  bf16
__global__ void __launch_bounds__(256)
prep_h_kernel(const float* __restrict__ hs, const float* __restrict__ pe,
              __bf16* __restrict__ h, int n) {
    int i = blockIdx.x * blockDim.x + threadIdx.x;
    if (i < n) h[i] = (__bf16)(hs[i] + pe[i]);
}

// W [256, N] (row-major, f32)  ->  Wt [N, 256] (row-major, bf16)
__global__ void __launch_bounds__(256)
convert_wt_kernel(const float* __restrict__ W, __bf16* __restrict__ Wt,
                  int N, int total) {
    int i = blockIdx.x * blockDim.x + threadIdx.x;
    if (i < total) {
        int k = i / N, n2 = i % N;
        Wt[(size_t)n2 * 256 + k] = (__bf16)W[(size_t)k * N + n2];
    }
}

// ---------------------------------------------------------------------------
// C[M,N] = A[M,256] @ Bt[N,256]^T + bias ; one 16x16 tile per wave,
// 8x v_wmma_f32_16x16x32_bf16 over K=256.
__global__ void __launch_bounds__(128)
gemm_bf16_wmma(const __bf16* __restrict__ A, const __bf16* __restrict__ Bt,
               const float* __restrict__ bias, float* __restrict__ C,
               int Ncols, int nColTiles) {
    const int wid  = threadIdx.x >> 5;
    const int lane = threadIdx.x & 31;
    const int rowTile = blockIdx.x;
    const int colTile = blockIdx.y * 4 + wid;
    if (colTile >= nColTiles) return;          // wave-uniform: EXEC stays all-1s

    const int hi  = lane >> 4;                 // half-wave select
    const int ln  = lane & 15;
    const int row = rowTile * 16 + ln;         // A fragment: M = lane&15
    const int col = colTile * 16 + ln;         // B/C fragment: N = lane&15

    const __bf16* arow = A  + (size_t)row * 256;
    const __bf16* bcol = Bt + (size_t)col * 256;

    v8f c = {};
#pragma unroll
    for (int kb = 0; kb < 256; kb += 32) {
        // A 16x32 bf16 fragment: element e -> K = e + 16*hi
        v8bf alo = *(const v8bf*)(arow + kb + 16 * hi);       // K = kb+16hi .. +7
        v8bf ahi = *(const v8bf*)(arow + kb + 16 * hi + 8);   // K = kb+16hi+8 ..
        v16bf a;
#pragma unroll
        for (int i = 0; i < 8; ++i) { a[i] = alo[i]; a[i + 8] = ahi[i]; }
        // B 32x16 bf16 fragment: lane = column N, element e -> K = e + 16*hi
        v16bf b = *(const v16bf*)(bcol + kb + 16 * hi);
        c = __builtin_amdgcn_wmma_f32_16x16x32_bf16(
                false, a, false, b, (short)0, c, false, false);
    }

    const float bv = bias[col];
#pragma unroll
    for (int r = 0; r < 8; ++r) {              // C/D: VGPR r -> M = r + 8*hi
        int mrow = rowTile * 16 + r + 8 * hi;
        C[(size_t)mrow * Ncols + col] = c[r] + bv;
    }
}

// ---------------------------------------------------------------------------
// One wave per (n, q, m); lane = channel (0..31).
// softmax over 12 logits, then 12-point bilinear sampling, write bf16 output.
__global__ void __launch_bounds__(256)
deform_sample_kernel(const float* __restrict__ values,   // [ROWS, 8, 32]
                     const float* __restrict__ off,      // [ROWS, 192]
                     const float* __restrict__ attn,     // [ROWS, 96]
                     const float* __restrict__ refp,     // [ROWS, 3, 2]
                     __bf16* __restrict__ t)             // [ROWS, 256]
{
    const int lane = threadIdx.x & 31;
    const int w    = (blockIdx.x * blockDim.x + threadIdx.x) >> 5;
    const int m    = w & 7;
    const int qn   = w >> 3;                   // n*NQ + q
    const int nIdx = qn / NQ;

    // softmax over L*K = 12 logits (redundant per-lane, all hit same lines)
    const float* aw = attn + (size_t)qn * 96 + m * 12;
    float lg[12];
    float mx = -1e30f;
#pragma unroll
    for (int i = 0; i < 12; ++i) { lg[i] = aw[i]; mx = fmaxf(mx, lg[i]); }
    float s = 0.f;
#pragma unroll
    for (int i = 0; i < 12; ++i) { lg[i] = __expf(lg[i] - mx); s += lg[i]; }
    const float inv = 1.f / s;

    const float* offp = off + (size_t)qn * 192 + m * 24;
    const float* rp   = refp + (size_t)qn * 6;

    const int Hs[3] = {96, 48, 24}, Ws[3] = {96, 48, 24};
    const int starts[3] = {0, 9216, 11520};

    float acc = 0.f;
#pragma unroll
    for (int l = 0; l < NLVL; ++l) {
        const int   Hi = Hs[l], Wi = Ws[l];
        const float Hf = (float)Hi, Wf = (float)Wi;
        const float ry = rp[l * 2 + 0], rx = rp[l * 2 + 1];
        const float* lv = values +
            (((size_t)nIdx * NQ + starts[l]) * MHEADS + m) * DHEAD + lane;
#pragma unroll
        for (int k = 0; k < NPTS; ++k) {
            const float oy = offp[l * 8 + k * 2 + 0];
            const float ox = offp[l * 8 + k * 2 + 1];
            // grid x = ref_x + off_x/W, grid y = ref_y + off_y/H  (post-reversal)
            const float x = ((rx + ox / Wf + 1.f) * Wf - 1.f) * 0.5f;
            const float y = ((ry + oy / Hf + 1.f) * Hf - 1.f) * 0.5f;
            const float x0f = floorf(x), y0f = floorf(y);
            const int x0 = (int)x0f, y0 = (int)y0f, x1 = x0 + 1, y1 = y0 + 1;
            const float wx1 = x - x0f, wx0 = 1.f - wx1;
            const float wy1 = y - y0f, wy0 = 1.f - wy1;
            const bool vx0 = (x0 >= 0) & (x0 < Wi), vx1 = (x1 >= 0) & (x1 < Wi);
            const bool vy0 = (y0 >= 0) & (y0 < Hi), vy1 = (y1 >= 0) & (y1 < Hi);
            float v00 = 0.f, v10 = 0.f, v01 = 0.f, v11 = 0.f;
            if (vx0 & vy0) v00 = lv[(size_t)(y0 * Wi + x0) * (MHEADS * DHEAD)];
            if (vx1 & vy0) v10 = lv[(size_t)(y0 * Wi + x1) * (MHEADS * DHEAD)];
            if (vx0 & vy1) v01 = lv[(size_t)(y1 * Wi + x0) * (MHEADS * DHEAD)];
            if (vx1 & vy1) v11 = lv[(size_t)(y1 * Wi + x1) * (MHEADS * DHEAD)];
            const float bil = v00 * wx0 * wy0 + v10 * wx1 * wy0 +
                              v01 * wx0 * wy1 + v11 * wx1 * wy1;
            acc += lg[l * NPTS + k] * inv * bil;
        }
    }
    t[((size_t)qn * MHEADS + m) * DHEAD + lane] = (__bf16)acc;
}

// ---------------------------------------------------------------------------
extern "C" void kernel_launch(void* const* d_in, const int* in_sizes, int n_in,
                              void* d_out, int out_size, void* d_ws, size_t ws_size,
                              hipStream_t stream) {
    const float* hs     = (const float*)d_in[0];
    const float* pe     = (const float*)d_in[1];
    const float* refp   = (const float*)d_in[2];
    const float* W_v    = (const float*)d_in[3];
    const float* b_v    = (const float*)d_in[4];
    const float* W_off  = (const float*)d_in[5];
    const float* b_off  = (const float*)d_in[6];
    const float* W_attn = (const float*)d_in[7];
    const float* b_attn = (const float*)d_in[8];
    const float* W_o    = (const float*)d_in[9];
    const float* b_o    = (const float*)d_in[10];
    float* out = (float*)d_out;

    char* ws = (char*)d_ws;
    size_t cur = 0;
    auto take = [&](size_t bytes) -> char* {
        char* p = ws + cur;
        cur += (bytes + 255) & ~(size_t)255;
        return p;
    };
    __bf16* h_bf    = (__bf16*)take((size_t)ROWS * 256 * 2);
    __bf16* wv_t    = (__bf16*)take((size_t)256 * 256 * 2);
    __bf16* woff_t  = (__bf16*)take((size_t)192 * 256 * 2);
    __bf16* wattn_t = (__bf16*)take((size_t)96 * 256 * 2);
    __bf16* wo_t    = (__bf16*)take((size_t)256 * 256 * 2);
    float*  val     = (float*)take((size_t)ROWS * 256 * 4);
    float*  offb    = (float*)take((size_t)ROWS * 192 * 4);
    float*  attnb   = (float*)take((size_t)ROWS * 96 * 4);
    __bf16* t_bf    = (__bf16*)take((size_t)ROWS * 256 * 2);

    // weight transpose + bf16 convert
    convert_wt_kernel<<<(256 * 256 + 255) / 256, 256, 0, stream>>>(W_v, wv_t, 256, 256 * 256);
    convert_wt_kernel<<<(256 * 192 + 255) / 256, 256, 0, stream>>>(W_off, woff_t, 192, 256 * 192);
    convert_wt_kernel<<<(256 * 96 + 255) / 256, 256, 0, stream>>>(W_attn, wattn_t, 96, 256 * 96);
    convert_wt_kernel<<<(256 * 256 + 255) / 256, 256, 0, stream>>>(W_o, wo_t, 256, 256 * 256);

    // h = hidden + pos (bf16)
    const int nelem = ROWS * 256;
    prep_h_kernel<<<(nelem + 255) / 256, 256, 0, stream>>>(hs, pe, h_bf, nelem);

    // projections via WMMA
    dim3 blk(128);
    gemm_bf16_wmma<<<dim3(ROWS / 16, (16 + 3) / 4), blk, 0, stream>>>(h_bf, wv_t,    b_v,    val,   256, 16);
    gemm_bf16_wmma<<<dim3(ROWS / 16, (12 + 3) / 4), blk, 0, stream>>>(h_bf, woff_t,  b_off,  offb,  192, 12);
    gemm_bf16_wmma<<<dim3(ROWS / 16, (6  + 3) / 4), blk, 0, stream>>>(h_bf, wattn_t, b_attn, attnb,  96,  6);

    // softmax + deformable bilinear sampling: ROWS*8 waves, 8 waves/block
    deform_sample_kernel<<<ROWS, 256, 0, stream>>>(val, offb, attnb, refp, t_bf);

    // output projection via WMMA -> d_out (f32)
    gemm_bf16_wmma<<<dim3(ROWS / 16, 4), blk, 0, stream>>>(t_bf, wo_t, b_o, out, 256, 16);
}
